// EGNNDiffusionModel_1812476199212
// MI455X (gfx1250) — matrix operand order repaired and checked by reading, not compiled, not run on hardware
//
#include <hip/hip_runtime.h>

typedef _Float16 f16;
typedef __attribute__((ext_vector_type(16))) _Float16 v16h;
typedef __attribute__((ext_vector_type(8)))  _Float16 v8h;
typedef __attribute__((ext_vector_type(8)))  float    v8f;

// ---------------------------------------------------------------------------
// WMMA wrapper: D = A(16x32 f16) * B(32x16 f16) + C(16x16 f32)
// ---------------------------------------------------------------------------
__device__ __forceinline__ v8f wmma_f16(v16h a, v16h b, v8f c) {
  return __builtin_amdgcn_wmma_f32_16x16x32_f16(false, a, false, b, (short)0, c,
                                                false, false);
}

// ---------------------------------------------------------------------------
// A-fragment (16x32 f16) from an f16 row (global or LDS).
// CDNA5 16-bit A layout: lane<16 -> elems 0..7 = K+0..7,  8..15 = K+16..23
//                        lane>=16-> elems 0..7 = K+8..15, 8..15 = K+24..31
// ---------------------------------------------------------------------------
__device__ __forceinline__ v16h load_a_frag(const f16* row, int lane) {
  const int b = (lane < 16) ? 0 : 8;
  v8h lo = *reinterpret_cast<const v8h*>(row + b);
  v8h hi = *reinterpret_cast<const v8h*>(row + b + 16);
  return __builtin_shufflevector(lo, hi, 0, 1, 2, 3, 4, 5, 6, 7, 8, 9, 10, 11,
                                 12, 13, 14, 15);
}

// ---------------------------------------------------------------------------
// Pack a [K, Ncols] f32 row-major matrix into f16 B-fragments.
// Fragment (ks, nt) covers K rows [ks*32, ks*32+32), cols [nt*16, nt*16+16).
// Element (frag, lane, j): col = nt*16 + (lane&15),
//                          krow = ks*32 + (lane<16 ? 0:8) + (j&7) + (j>=8 ? 16:0).
// cols >= Ncols are zero-filled (pads bp_w2 from 4 to 16 cols).
// ---------------------------------------------------------------------------
__global__ void pack_b_frags(const float* __restrict__ W, f16* __restrict__ out,
                             int Kblocks, int Ntiles, int Ncols, int ldw) {
  int idx = blockIdx.x * blockDim.x + threadIdx.x;
  int total = Kblocks * Ntiles * 512;
  if (idx >= total) return;
  int j    = idx & 15;
  int lane = (idx >> 4) & 31;
  int frag = idx >> 9;
  int nt   = frag % Ntiles;
  int ks   = frag / Ntiles;
  int col  = (nt << 4) + (lane & 15);
  int krow = (ks << 5) + ((lane >> 4) << 3) + (j & 7) + ((j >> 3) << 4);
  float v = (col < Ncols) ? W[(size_t)krow * ldw + col] : 0.0f;
  out[idx] = (f16)v;
}

// ---------------------------------------------------------------------------
// Time embedding: scalar t -> [32]
// ---------------------------------------------------------------------------
__global__ void time_embed_kernel(const int* __restrict__ t,
                                  const float* __restrict__ w1,
                                  const float* __restrict__ b1,
                                  const float* __restrict__ w2,
                                  const float* __restrict__ b2,
                                  float* __restrict__ temb) {
  int j = threadIdx.x;
  if (j >= 32) return;
  float t_in = (float)t[0] / 1000.0f;
  float acc = b2[j];
#pragma unroll 4
  for (int k = 0; k < 32; ++k)
    acc += fmaxf(t_in * w1[k] + b1[k], 0.0f) * w2[k * 32 + j];
  temb[j] = acc;
}

// x_input[n] = concat(x_t[n] (64), cond_embed (32), t_embed (32)); f32 + f16
__global__ void build_x_kernel(const float* __restrict__ x_t,
                               const float* __restrict__ cond,
                               const float* __restrict__ temb,
                               float* __restrict__ x_in, f16* __restrict__ x_inh,
                               int N) {
  int idx = blockIdx.x * blockDim.x + threadIdx.x;
  if (idx >= N * 128) return;
  int n = idx >> 7, f = idx & 127;
  float v;
  if (f < 64)      v = x_t[(size_t)n * 64 + f];
  else if (f < 96) v = cond[f - 64];
  else             v = temb[f - 96];
  x_in[idx] = v;
  x_inh[idx] = (f16)v;
}

__global__ void f32_to_f16_kernel(const float* __restrict__ in,
                                  f16* __restrict__ outh, int n) {
  int i = blockIdx.x * blockDim.x + threadIdx.x;
  if (i < n) outh[i] = (f16)in[i];
}

// ---------------------------------------------------------------------------
// EGNN edge-message kernel (one layer). Per wave: 32-edge tiles (2 subtiles)
// so every LDS B-fragment feeds two independent WMMA chains.
//   h = [x[dst] | x[src] | dist] ; msg = relu(h@W1+b1)@W2+b2 -> atomic xout[dst]
// ---------------------------------------------------------------------------
__global__ __launch_bounds__(256)
void egnn_feat_kernel(const f16* __restrict__ xh, const float* __restrict__ pos,
                      const int* __restrict__ ei, int E, int N,
                      const f16* __restrict__ w1p, const f16* __restrict__ w2p,
                      const float* __restrict__ w1drow,
                      const float* __restrict__ b1, const float* __restrict__ b2,
                      float* __restrict__ xout) {
  extern __shared__ char smem[];
  f16*   w1s   = (f16*)smem;              // 8ks x 8nt x 512 = 32768 f16
  f16*   w2s   = w1s + 32768;             // 4ks x 8nt x 512 = 16384 f16
  f16*   ys    = w2s + 16384;             // 8 waves x 32x128 = 32768 f16
  int*   sidxs = (int*)(ys + 32768);      // 8 x 32
  int*   didxs = sidxs + 256;             // 8 x 32
  float* dists = (float*)(didxs + 256);   // 8 x 32

  for (int i = threadIdx.x; i < 32768; i += 256) w1s[i] = w1p[i];
  for (int i = threadIdx.x; i < 16384; i += 256) w2s[i] = w2p[i];
  __syncthreads();

  const int lane = threadIdx.x & 31;
  const int wv   = threadIdx.x >> 5;
  const int ml   = lane & 15;
  const int hi8  = (lane < 16) ? 0 : 8;
  const int wv32 = wv * 32;
  f16* ysw = ys + wv * 4096;

  // loop-invariant per-lane bias / dist-row registers (col = nt*16 + ml)
  float b1r[8], wdr[8], b2r[8];
#pragma unroll
  for (int nt = 0; nt < 8; ++nt) {
    b1r[nt] = b1[(nt << 4) + ml];
    wdr[nt] = w1drow[(nt << 4) + ml];
    b2r[nt] = b2[(nt << 4) + ml];
  }

  const int M = E + N;
  const int nTiles = (M + 31) >> 5;
  const int waveId = (blockIdx.x * blockDim.x + threadIdx.x) >> 5;
  const int nWaves = (gridDim.x * blockDim.x) >> 5;

  for (int tile = waveId; tile < nTiles; tile += nWaves) {
    const int base = tile << 5;
    {
      int e = base + lane;                 // all 32 lanes: one edge each
      int s = 0, d = 0;
      float dist = 0.0f;
      if (e < M) {
        if (e < E) { s = ei[e]; d = ei[E + e]; } else { s = e - E; d = s; }
        float ex = pos[s * 3 + 0] - pos[d * 3 + 0];
        float ey = pos[s * 3 + 1] - pos[d * 3 + 1];
        float ez = pos[s * 3 + 2] - pos[d * 3 + 2];
        dist = sqrtf(ex * ex + ey * ey + ez * ez + 1e-8f);
      }
      sidxs[wv32 + lane] = s;
      didxs[wv32 + lane] = d;
      dists[wv32 + lane] = dist;
    }
    __builtin_amdgcn_wave_barrier();

    const int s0 = sidxs[wv32 + ml],      d0 = didxs[wv32 + ml];
    const int s1 = sidxs[wv32 + 16 + ml], d1 = didxs[wv32 + 16 + ml];

    // A fragments: K 0..127 = x[dst], 128..255 = x[src]; two subtiles
    v16h a0[8], a1[8];
#pragma unroll
    for (int ks = 0; ks < 4; ++ks) {
      a0[ks]     = load_a_frag(xh + (size_t)d0 * 128 + ks * 32, lane);
      a0[4 + ks] = load_a_frag(xh + (size_t)s0 * 128 + ks * 32, lane);
      a1[ks]     = load_a_frag(xh + (size_t)d1 * 128 + ks * 32, lane);
      a1[4 + ks] = load_a_frag(xh + (size_t)s1 * 128 + ks * 32, lane);
    }

    // dist values for the C-layout rows of each subtile (vector LDS loads)
    float4 dA0 = *(const float4*)(dists + wv32 + hi8);
    float4 dA1 = *(const float4*)(dists + wv32 + hi8 + 4);
    float4 dB0 = *(const float4*)(dists + wv32 + 16 + hi8);
    float4 dB1 = *(const float4*)(dists + wv32 + 16 + hi8 + 4);
    float dr0[8] = {dA0.x, dA0.y, dA0.z, dA0.w, dA1.x, dA1.y, dA1.z, dA1.w};
    float dr1[8] = {dB0.x, dB0.y, dB0.z, dB0.w, dB1.x, dB1.y, dB1.z, dB1.w};

    // stage 1: h @ W1 + b1 (+ dist rank-1 term), relu -> LDS Y (32x128 f16)
    for (int nt = 0; nt < 8; ++nt) {
      v8f acc0, acc1;
#pragma unroll
      for (int v = 0; v < 8; ++v) {
        acc0[v] = b1r[nt] + dr0[v] * wdr[nt];
        acc1[v] = b1r[nt] + dr1[v] * wdr[nt];
      }
#pragma unroll
      for (int ks = 0; ks < 8; ++ks) {
        v16h bf = *(const v16h*)(w1s + (((ks << 3) + nt) << 9) + (lane << 4));
        acc0 = wmma_f16(a0[ks], bf, acc0);
        acc1 = wmma_f16(a1[ks], bf, acc1);
      }
      const int col = (nt << 4) + ml;
#pragma unroll
      for (int v = 0; v < 8; ++v) {
        ysw[(v + hi8) * 128 + col]      = (f16)fmaxf(acc0[v], 0.0f);
        ysw[(16 + v + hi8) * 128 + col] = (f16)fmaxf(acc1[v], 0.0f);
      }
    }
    __builtin_amdgcn_wave_barrier();

    v16h ya0[4], ya1[4];
#pragma unroll
    for (int k2 = 0; k2 < 4; ++k2) {
      ya0[k2] = load_a_frag(ysw + ml * 128 + k2 * 32, lane);
      ya1[k2] = load_a_frag(ysw + (16 + ml) * 128 + k2 * 32, lane);
    }

    // dst indices for scatter rows (vector LDS loads)
    int4 i0a = *(const int4*)(didxs + wv32 + hi8);
    int4 i0b = *(const int4*)(didxs + wv32 + hi8 + 4);
    int4 i1a = *(const int4*)(didxs + wv32 + 16 + hi8);
    int4 i1b = *(const int4*)(didxs + wv32 + 16 + hi8 + 4);
    int dR0[8] = {i0a.x, i0a.y, i0a.z, i0a.w, i0b.x, i0b.y, i0b.z, i0b.w};
    int dR1[8] = {i1a.x, i1a.y, i1a.z, i1a.w, i1b.x, i1b.y, i1b.z, i1b.w};

    // stage 2: Y @ W2 + b2, scatter-add into xout[dst]
    for (int nt = 0; nt < 8; ++nt) {
      v8f acc0, acc1;
#pragma unroll
      for (int v = 0; v < 8; ++v) { acc0[v] = b2r[nt]; acc1[v] = b2r[nt]; }
#pragma unroll
      for (int k2 = 0; k2 < 4; ++k2) {
        v16h bf = *(const v16h*)(w2s + (((k2 << 3) + nt) << 9) + (lane << 4));
        acc0 = wmma_f16(ya0[k2], bf, acc0);
        acc1 = wmma_f16(ya1[k2], bf, acc1);
      }
      const int col = (nt << 4) + ml;
#pragma unroll
      for (int v = 0; v < 8; ++v) {
        int r0 = base + v + hi8;
        if (r0 < M) atomicAdd(xout + (size_t)dR0[v] * 128 + col, acc0[v]);
        int r1 = base + 16 + v + hi8;
        if (r1 < M) atomicAdd(xout + (size_t)dR1[v] * 128 + col, acc1[v]);
      }
    }
    __builtin_amdgcn_wave_barrier();
  }
}

// ---------------------------------------------------------------------------
// Coordinate update: pos_out[dst] += (relu(dist@cw1+cb1)@cw2+cb2) * edge_vec
// ---------------------------------------------------------------------------
__global__ void coord_kernel(const float* __restrict__ pos,
                             const int* __restrict__ ei, int E, int N,
                             const float* __restrict__ w1,
                             const float* __restrict__ b1,
                             const float* __restrict__ w2,
                             const float* __restrict__ b2,
                             float* __restrict__ pos_out) {
  int e = blockIdx.x * blockDim.x + threadIdx.x;
  int M = E + N;
  if (e >= M) return;
  int s, d;
  if (e < E) { s = ei[e]; d = ei[E + e]; } else { s = e - E; d = s; }
  float ex = pos[s * 3 + 0] - pos[d * 3 + 0];
  float ey = pos[s * 3 + 1] - pos[d * 3 + 1];
  float ez = pos[s * 3 + 2] - pos[d * 3 + 2];
  float dist = sqrtf(ex * ex + ey * ey + ez * ez + 1e-8f);
  float out = b2[0];
#pragma unroll 4
  for (int k = 0; k < 128; ++k)
    out += fmaxf(dist * w1[k] + b1[k], 0.0f) * w2[k];
  atomicAdd(pos_out + (size_t)d * 3 + 0, out * ex);
  atomicAdd(pos_out + (size_t)d * 3 + 1, out * ey);
  atomicAdd(pos_out + (size_t)d * 3 + 2, out * ez);
}

// out[:, :64] slice of x2
__global__ void out_head_kernel(const float* __restrict__ x2,
                                float* __restrict__ out, int N) {
  int idx = blockIdx.x * blockDim.x + threadIdx.x;
  if (idx >= N * 64) return;
  int n = idx >> 6, f = idx & 63;
  out[idx] = x2[(size_t)n * 128 + f];
}

// ---------------------------------------------------------------------------
// Bond predictor: relu([x2[src]|x2[dst]] @ W1 + b1) @ W2 + b2 -> [E,4]
// Same 32-edge-per-wave two-stage WMMA pattern; W2 zero-padded 64x16.
// ---------------------------------------------------------------------------
__global__ __launch_bounds__(256)
void bond_kernel(const f16* __restrict__ xh, const int* __restrict__ ei, int E,
                 const f16* __restrict__ w1p, const f16* __restrict__ w2p,
                 const float* __restrict__ b1, const float* __restrict__ b2,
                 float* __restrict__ out) {
  extern __shared__ char smem[];
  f16* w1s   = (f16*)smem;              // 8ks x 4nt x 512 = 16384 f16
  f16* w2s   = w1s + 16384;             // 2ks x 1nt x 512 = 1024 f16
  f16* ys    = w2s + 1024;              // 8 waves x 32x64 = 16384 f16
  int* sidxs = (int*)(ys + 16384);      // 8 x 32
  int* didxs = sidxs + 256;             // 8 x 32

  for (int i = threadIdx.x; i < 16384; i += 256) w1s[i] = w1p[i];
  for (int i = threadIdx.x; i < 1024;  i += 256) w2s[i] = w2p[i];
  __syncthreads();

  const int lane = threadIdx.x & 31;
  const int wv   = threadIdx.x >> 5;
  const int ml   = lane & 15;
  const int hi8  = (lane < 16) ? 0 : 8;
  const int wv32 = wv * 32;
  f16* ysw = ys + wv * 2048;

  float b1r[4];
#pragma unroll
  for (int nt = 0; nt < 4; ++nt) b1r[nt] = b1[(nt << 4) + ml];
  const float b2r = (ml < 4) ? b2[ml] : 0.0f;

  const int nTiles = (E + 31) >> 5;
  const int waveId = (blockIdx.x * blockDim.x + threadIdx.x) >> 5;
  const int nWaves = (gridDim.x * blockDim.x) >> 5;

  for (int tile = waveId; tile < nTiles; tile += nWaves) {
    const int base = tile << 5;
    {
      int e = base + lane;
      int s = 0, d = 0;
      if (e < E) { s = ei[e]; d = ei[E + e]; }
      sidxs[wv32 + lane] = s;
      didxs[wv32 + lane] = d;
    }
    __builtin_amdgcn_wave_barrier();

    const int s0 = sidxs[wv32 + ml],      d0 = didxs[wv32 + ml];
    const int s1 = sidxs[wv32 + 16 + ml], d1 = didxs[wv32 + 16 + ml];

    // K 0..127 = x2[src], 128..255 = x2[dst]
    v16h a0[8], a1[8];
#pragma unroll
    for (int ks = 0; ks < 4; ++ks) {
      a0[ks]     = load_a_frag(xh + (size_t)s0 * 128 + ks * 32, lane);
      a0[4 + ks] = load_a_frag(xh + (size_t)d0 * 128 + ks * 32, lane);
      a1[ks]     = load_a_frag(xh + (size_t)s1 * 128 + ks * 32, lane);
      a1[4 + ks] = load_a_frag(xh + (size_t)d1 * 128 + ks * 32, lane);
    }

    // stage 1: [.,256] @ [256,64] + b1, relu -> Y (32x64 f16)
    for (int nt = 0; nt < 4; ++nt) {
      v8f acc0, acc1;
#pragma unroll
      for (int v = 0; v < 8; ++v) { acc0[v] = b1r[nt]; acc1[v] = b1r[nt]; }
#pragma unroll
      for (int ks = 0; ks < 8; ++ks) {
        v16h bf = *(const v16h*)(w1s + (((ks << 2) + nt) << 9) + (lane << 4));
        acc0 = wmma_f16(a0[ks], bf, acc0);
        acc1 = wmma_f16(a1[ks], bf, acc1);
      }
      const int col = (nt << 4) + ml;
#pragma unroll
      for (int v = 0; v < 8; ++v) {
        ysw[(v + hi8) * 64 + col]      = (f16)fmaxf(acc0[v], 0.0f);
        ysw[(16 + v + hi8) * 64 + col] = (f16)fmaxf(acc1[v], 0.0f);
      }
    }
    __builtin_amdgcn_wave_barrier();

    v16h ya0[2], ya1[2];
#pragma unroll
    for (int k2 = 0; k2 < 2; ++k2) {
      ya0[k2] = load_a_frag(ysw + ml * 64 + k2 * 32, lane);
      ya1[k2] = load_a_frag(ysw + (16 + ml) * 64 + k2 * 32, lane);
    }

    // stage 2: [.,64] @ [64,16(pad)] + b2
    {
      v8f acc0, acc1;
#pragma unroll
      for (int v = 0; v < 8; ++v) { acc0[v] = b2r; acc1[v] = b2r; }
#pragma unroll
      for (int k2 = 0; k2 < 2; ++k2) {
        v16h bf = *(const v16h*)(w2s + (k2 << 9) + (lane << 4));
        acc0 = wmma_f16(ya0[k2], bf, acc0);
        acc1 = wmma_f16(ya1[k2], bf, acc1);
      }
#pragma unroll
      for (int v = 0; v < 8; ++v) {
        int r0 = base + v + hi8;
        if (ml < 4 && r0 < E) out[(size_t)r0 * 4 + ml] = acc0[v];
        int r1 = base + 16 + v + hi8;
        if (ml < 4 && r1 < E) out[(size_t)r1 * 4 + ml] = acc1[v];
      }
    }
    __builtin_amdgcn_wave_barrier();
  }
}

// ---------------------------------------------------------------------------
// Host-side orchestration
// ---------------------------------------------------------------------------
extern "C" void kernel_launch(void* const* d_in, const int* in_sizes, int n_in,
                              void* d_out, int out_size, void* d_ws, size_t ws_size,
                              hipStream_t stream) {
  const float* x_t      = (const float*)d_in[0];
  const float* pos      = (const float*)d_in[1];
  const int*   ei       = (const int*)  d_in[2];
  const int*   t        = (const int*)  d_in[3];
  const float* cond     = (const float*)d_in[4];
  const float* te_w1    = (const float*)d_in[5];
  const float* te_b1    = (const float*)d_in[6];
  const float* te_w2    = (const float*)d_in[7];
  const float* te_b2    = (const float*)d_in[8];
  const float* l1_nm_w1 = (const float*)d_in[9];
  const float* l1_nm_b1 = (const float*)d_in[10];
  const float* l1_nm_w2 = (const float*)d_in[11];
  const float* l1_nm_b2 = (const float*)d_in[12];
  const float* l1_cm_w1 = (const float*)d_in[13];
  const float* l1_cm_b1 = (const float*)d_in[14];
  const float* l1_cm_w2 = (const float*)d_in[15];
  const float* l1_cm_b2 = (const float*)d_in[16];
  const float* l2_nm_w1 = (const float*)d_in[17];
  const float* l2_nm_b1 = (const float*)d_in[18];
  const float* l2_nm_w2 = (const float*)d_in[19];
  const float* l2_nm_b2 = (const float*)d_in[20];
  const float* l2_cm_w1 = (const float*)d_in[21];
  const float* l2_cm_b1 = (const float*)d_in[22];
  const float* l2_cm_w2 = (const float*)d_in[23];
  const float* l2_cm_b2 = (const float*)d_in[24];
  const float* bp_w1    = (const float*)d_in[25];
  const float* bp_b1    = (const float*)d_in[26];
  const float* bp_w2    = (const float*)d_in[27];
  const float* bp_b2    = (const float*)d_in[28];

  const int N = in_sizes[0] / 64;   // 50000
  const int E = in_sizes[2] / 2;    // 800000
  const int M = E + N;
  float* outf = (float*)d_out;

  // workspace carve (256B aligned regions)
  char* wsb = (char*)d_ws;
  size_t off = 0;
  auto carve = [&](size_t bytes) -> void* {
    void* p = wsb + off;
    off = (off + bytes + 255) & ~(size_t)255;
    return p;
  };
  float* x_in  = (float*)carve((size_t)N * 128 * 4);
  float* x1    = (float*)carve((size_t)N * 128 * 4);
  float* x2    = (float*)carve((size_t)N * 128 * 4);
  f16*   x_inh = (f16*)  carve((size_t)N * 128 * 2);
  f16*   x1h   = (f16*)  carve((size_t)N * 128 * 2);
  f16*   x2h   = (f16*)  carve((size_t)N * 128 * 2);
  float* pos1  = (float*)carve((size_t)N * 3 * 4);
  float* temb  = (float*)carve(32 * 4);
  f16* l1w1p = (f16*)carve(32768 * 2);
  f16* l1w2p = (f16*)carve(16384 * 2);
  f16* l2w1p = (f16*)carve(32768 * 2);
  f16* l2w2p = (f16*)carve(16384 * 2);
  f16* bpw1p = (f16*)carve(16384 * 2);
  f16* bpw2p = (f16*)carve(1024 * 2);

  // --- pack weights to f16 B-fragments (reused for every edge) ---
  pack_b_frags<<<128, 256, 0, stream>>>(l1_nm_w1, l1w1p, 8, 8, 128, 128);
  pack_b_frags<<<64, 256, 0, stream>>>(l1_nm_w2, l1w2p, 4, 8, 128, 128);
  pack_b_frags<<<128, 256, 0, stream>>>(l2_nm_w1, l2w1p, 8, 8, 128, 128);
  pack_b_frags<<<64, 256, 0, stream>>>(l2_nm_w2, l2w2p, 4, 8, 128, 128);
  pack_b_frags<<<64, 256, 0, stream>>>(bp_w1, bpw1p, 8, 4, 64, 64);
  pack_b_frags<<<4, 256, 0, stream>>>(bp_w2, bpw2p, 2, 1, 4, 4);

  // --- input assembly ---
  time_embed_kernel<<<1, 32, 0, stream>>>(t, te_w1, te_b1, te_w2, te_b2, temb);
  build_x_kernel<<<(N * 128 + 255) / 256, 256, 0, stream>>>(x_t, cond, temb,
                                                            x_in, x_inh, N);
  hipMemcpyAsync(x1, x_in, (size_t)N * 128 * 4, hipMemcpyDeviceToDevice, stream);
  hipMemcpyAsync(pos1, pos, (size_t)N * 3 * 4, hipMemcpyDeviceToDevice, stream);

  const size_t FEAT_SMEM = 32768 * 2 + 16384 * 2 + 32768 * 2 + 3 * 256 * 4;
  const size_t BOND_SMEM = 16384 * 2 + 1024 * 2 + 16384 * 2 + 2 * 256 * 4;

  int featTiles = (M + 31) >> 5;
  int featBlocks = (featTiles + 7) / 8;
  if (featBlocks > 2048) featBlocks = 2048;
  int bondTiles = (E + 31) >> 5;
  int bondBlocks = (bondTiles + 7) / 8;
  if (bondBlocks > 2048) bondBlocks = 2048;

  // --- EGNN layer 1 ---
  egnn_feat_kernel<<<featBlocks, 256, FEAT_SMEM, stream>>>(
      x_inh, pos, ei, E, N, l1w1p, l1w2p, l1_nm_w1 + 256 * 128, l1_nm_b1,
      l1_nm_b2, x1);
  coord_kernel<<<(M + 255) / 256, 256, 0, stream>>>(
      pos, ei, E, N, l1_cm_w1, l1_cm_b1, l1_cm_w2, l1_cm_b2, pos1);
  f32_to_f16_kernel<<<(N * 128 + 255) / 256, 256, 0, stream>>>(x1, x1h, N * 128);

  // --- EGNN layer 2 (pos2 is dead: outputs only need x2) ---
  hipMemcpyAsync(x2, x1, (size_t)N * 128 * 4, hipMemcpyDeviceToDevice, stream);
  egnn_feat_kernel<<<featBlocks, 256, FEAT_SMEM, stream>>>(
      x1h, pos1, ei, E, N, l2w1p, l2w2p, l2_nm_w1 + 256 * 128, l2_nm_b1,
      l2_nm_b2, x2);
  f32_to_f16_kernel<<<(N * 128 + 255) / 256, 256, 0, stream>>>(x2, x2h, N * 128);

  // --- outputs ---
  out_head_kernel<<<(N * 64 + 255) / 256, 256, 0, stream>>>(x2, outf, N);
  bond_kernel<<<bondBlocks, 256, BOND_SMEM, stream>>>(
      x2h, ei, E, bpw1p, bpw2p, bp_b1, bp_b2, outf + (size_t)N * 64);
}
